// StepLayer_24206435680900
// MI455X (gfx1250) — matrix operand order repaired and checked by reading, not compiled, not run on hardware
//
#include <hip/hip_runtime.h>
#include <math.h>

#define CH    1024
#define PIVOT 16384
#define BK    32
#define NK    (CH / BK)          // 32 k-steps
#define BM    256
#define BN    128
#define LDSW  40                 // bf16 elems per LDS row: 32 data + 8 pad (80B, 16B aligned)
#define A_LDS_ELEMS (BM * LDSW)  // 10240
#define B_LDS_ELEMS (BN * LDSW)  // 5120

typedef __attribute__((ext_vector_type(16))) __bf16 v16bf;
typedef __attribute__((ext_vector_type(8)))  __bf16 v8bf;
typedef __attribute__((ext_vector_type(4)))  __bf16 v4bf;
typedef __attribute__((ext_vector_type(8)))  float  v8f;
typedef __attribute__((ext_vector_type(4)))  float  v4f;
typedef __attribute__((ext_vector_type(2)))  float  v2f;
typedef __attribute__((ext_vector_type(4)))  int    v4i;

#if __has_builtin(__builtin_amdgcn_global_load_async_to_lds_b128)
#define USE_ASYNC_LDS 1
typedef __attribute__((address_space(1))) v4i* gv4p;   // global (HIP prints as __device__)
typedef __attribute__((address_space(3))) v4i* lv4p;   // LDS
#endif

__device__ __forceinline__ void cp16(__bf16* lds, const __bf16* g) {
#if defined(USE_ASYNC_LDS)
  __builtin_amdgcn_global_load_async_to_lds_b128((gv4p)g, (lv4p)lds, 0, 0);
#else
  *(v8bf*)lds = *(const v8bf*)g;
#endif
}

__device__ __forceinline__ void wait_async0() {
#if defined(USE_ASYNC_LDS)
#if __has_builtin(__builtin_amdgcn_s_wait_asynccnt)
  __builtin_amdgcn_s_wait_asynccnt(0);
#else
  asm volatile("s_wait_asynccnt 0" ::: "memory");
#endif
#endif
}

__device__ __forceinline__ v16bf cat8(v8bf lo, v8bf hi) {
  return __builtin_shufflevector(lo, hi, 0,1,2,3,4,5,6,7,8,9,10,11,12,13,14,15);
}

// branchless tanh: copysign(1 - 2*rcp(e^{2|x|}+1), x); e->inf gives exactly 1
__device__ __forceinline__ float fast_tanh(float x) {
  float ax = fabsf(x);
  float e  = __expf(ax + ax);
  float t  = 1.0f - 2.0f * __builtin_amdgcn_rcpf(e + 1.0f);
  return copysignf(t, x);
}

// ---------------- prep: fp32 -> bf16 bulk convert (W and x block) ----------------
__global__ __launch_bounds__(256)
void cvt_f32_bf16(const float* __restrict__ src, __bf16* __restrict__ dst, size_t n4) {
  size_t i = (size_t)blockIdx.x * blockDim.x + threadIdx.x;
  if (i >= n4) return;
  v4f v = ((const v4f*)src)[i];
  v4bf o;
  o[0] = (__bf16)v[0]; o[1] = (__bf16)v[1];
  o[2] = (__bf16)v[2]; o[3] = (__bf16)v[3];
  ((v4bf*)dst)[i] = o;
}

// ---------------- prep: de-interleave Jx[p,j,k] -> Y[2p+k, j] (bf16) ----------------
__global__ __launch_bounds__(256)
void cvt_y(const float* __restrict__ xJ, __bf16* __restrict__ Y) {
  size_t t = (size_t)blockIdx.x * 256 + threadIdx.x;     // 0 .. 2*PIVOT*512-1
  int rr = (int)(t >> 9);                                // input row - PIVOT
  int cp = (int)(t & 511);                               // column pair index
  v2f v = *(const v2f*)(xJ + ((size_t)(PIVOT + rr) * CH + 2 * cp));
  int base = rr & ~1;                                    // 2p
  int col  = (rr & 1) * 512 + cp;                        // h*512 + cp
  Y[(size_t)(base + 0) * CH + col] = (__bf16)v.x;        // k = 0
  Y[(size_t)(base + 1) * CH + col] = (__bf16)v.y;        // k = 1
}

// ---------------- fused GEMM: C = A @ W^T, epilogue tanh(+bias) or Jacobian scatter ----------------
template <bool JAC>
__global__ __launch_bounds__(256)
void fused_gemm(const __bf16* __restrict__ A, const __bf16* __restrict__ Wb,
                const float* __restrict__ aux,    // !JAC: bias[CH]; JAC: hAx (d_out rows [0,PIVOT))
                float* __restrict__ out) {        // !JAC: rows [0,PIVOT); JAC: rows [PIVOT,3*PIVOT)
  __shared__ __bf16 As[2 * A_LDS_ELEMS];
  __shared__ __bf16 Bs[2 * B_LDS_ELEMS];

  const int tid  = threadIdx.x;
  const int lane = tid & 31;
  const int wave = tid >> 5;
  const int waveM = wave & 3;       // 4 waves * 64 rows = 256
  const int waveN = wave >> 2;      // 2 waves * 64 cols = 128
  const int lm = lane & 15;
  const int lh = lane >> 4;
  const int bm = blockIdx.x * BM;
  const int bn = blockIdx.y * BN;

  // staging mapping: A = one 64B row-chunk per thread; B = 32B half-row per thread
  const __bf16* agBase = A + (size_t)(bm + tid) * CH;
  const int brow = tid >> 1, bcol = (tid & 1) * 16;
  const __bf16* bgBase = Wb + (size_t)(bn + brow) * CH + bcol;
  __bf16* aLds = &As[tid * LDSW];
  __bf16* bLds = &Bs[brow * LDSW + bcol];

  auto stage = [&](int buf, int it) {
    const __bf16* ag = agBase + it * BK;
    __bf16* al = aLds + buf * A_LDS_ELEMS;
    cp16(al,      ag);
    cp16(al + 8,  ag + 8);
    cp16(al + 16, ag + 16);
    cp16(al + 24, ag + 24);
    const __bf16* bg = bgBase + it * BK;
    __bf16* bl = bLds + buf * B_LDS_ELEMS;
    cp16(bl,     bg);
    cp16(bl + 8, bg + 8);
  };

  v8f acc[4][4] = {};

  stage(0, 0);
  wait_async0();
  __syncthreads();

  for (int it = 0; it < NK; ++it) {
    const int cur = it & 1;
    if (it + 1 < NK) stage(cur ^ 1, it + 1);   // prefetch next tile into other buffer

    const __bf16* Ab = &As[cur * A_LDS_ELEMS];
    const __bf16* Bb = &Bs[cur * B_LDS_ELEMS];
    v16bf af[4], bfm[4];
    #pragma unroll
    for (int mt = 0; mt < 4; ++mt) {
      const __bf16* p = Ab + (waveM * 64 + mt * 16 + lm) * LDSW + lh * 8;
      af[mt] = cat8(*(const v8bf*)p, *(const v8bf*)(p + 16));
    }
    #pragma unroll
    for (int nt = 0; nt < 4; ++nt) {
      const __bf16* q = Bb + (waveN * 64 + nt * 16 + lm) * LDSW + lh * 8;
      bfm[nt] = cat8(*(const v8bf*)q, *(const v8bf*)(q + 16));
    }
    #pragma unroll
    for (int mt = 0; mt < 4; ++mt)
      #pragma unroll
      for (int nt = 0; nt < 4; ++nt)
        acc[mt][nt] = __builtin_amdgcn_wmma_f32_16x16x32_bf16(
            false, af[mt], false, bfm[nt], (short)0, acc[mt][nt], false, false);

    wait_async0();
    __syncthreads();
  }

  if constexpr (!JAC) {
    #pragma unroll
    for (int nt = 0; nt < 4; ++nt) {
      const int col = bn + waveN * 64 + nt * 16 + lm;
      const float bv = aux[col];
      #pragma unroll
      for (int mt = 0; mt < 4; ++mt) {
        #pragma unroll
        for (int r = 0; r < 8; ++r) {
          const int m = bm + waveM * 64 + mt * 16 + lh * 8 + r;
          out[(size_t)m * CH + col] = fast_tanh(acc[mt][nt][r] + bv);
        }
      }
    }
  } else {
    // Z[zr, i] -> out[(zr&~1) + (i>>9), 2*(i&511) + (zr&1)], scaled by 1 - hAx[zr>>1, i]^2
    #pragma unroll
    for (int nt = 0; nt < 4; ++nt) {
      const int i     = bn + waveN * 64 + nt * 16 + lm;
      const int h     = i >> 9;
      const int ocol0 = 2 * (i & 511);
      #pragma unroll
      for (int mt = 0; mt < 4; ++mt) {
        #pragma unroll
        for (int r = 0; r < 8; ++r) {
          const int zr = bm + waveM * 64 + mt * 16 + lh * 8 + r;
          const float o  = aux[(size_t)(zr >> 1) * CH + i];
          const float hp = 1.0f - o * o;
          out[(size_t)((zr & ~1) + h) * CH + (ocol0 + (zr & 1))] = hp * acc[mt][nt][r];
        }
      }
    }
  }
}

extern "C" void kernel_launch(void* const* d_in, const int* in_sizes, int n_in,
                              void* d_out, int out_size, void* d_ws, size_t ws_size,
                              hipStream_t stream) {
  (void)in_sizes; (void)n_in; (void)out_size; (void)ws_size;
  const float* xJ = (const float*)d_in[0];
  const float* W  = (const float*)d_in[1];
  const float* b  = (const float*)d_in[2];
  float* out = (float*)d_out;

  __bf16* Wb = (__bf16*)d_ws;                                  // 2 MB
  __bf16* Ab = (__bf16*)((char*)d_ws + (size_t)CH * CH * 2);   // 96 MB (x rows, then Y rows)

  cvt_f32_bf16<<<(CH * CH / 4) / 256, 256, 0, stream>>>(W, Wb, (size_t)CH * CH / 4);
  cvt_f32_bf16<<<((size_t)PIVOT * CH / 4) / 256, 256, 0, stream>>>(xJ, Ab, (size_t)PIVOT * CH / 4);
  cvt_y<<<((size_t)2 * PIVOT * 512) / 256, 256, 0, stream>>>(xJ, Ab + (size_t)PIVOT * CH);

  fused_gemm<false><<<dim3(PIVOT / BM, CH / BN), 256, 0, stream>>>(Ab, Wb, b, out);
  fused_gemm<true><<<dim3(2 * PIVOT / BM, CH / BN), 256, 0, stream>>>(
      Ab + (size_t)PIVOT * CH, Wb, out, out + (size_t)PIVOT * CH);
}
